// NAGphormer_12567074308299
// MI455X (gfx1250) — matrix operand order repaired and checked by reading, not compiled, not run on hardware
//
#include <hip/hip_runtime.h>

#define DEVI __device__ __forceinline__

typedef __attribute__((ext_vector_type(16))) __bf16 bf16x16;
typedef __attribute__((ext_vector_type(8)))  float  f32x8;
typedef __attribute__((ext_vector_type(4)))  unsigned int u32x4;
typedef __attribute__((ext_vector_type(8)))  int    i32x8;
typedef __attribute__((ext_vector_type(4)))  int    i32x4;

#if defined(__has_builtin)
#  if __has_builtin(__builtin_amdgcn_tensor_load_to_lds) && \
      __has_builtin(__builtin_amdgcn_s_wait_tensorcnt)
#    define USE_TDM 1
#  else
#    define USE_TDM 0
#  endif
#else
#  define USE_TDM 0
#endif

namespace {

constexpr int NN  = 100000;   // nodes
constexpr int EE  = 1600000;  // edges
constexpr int H   = 128;
constexpr int S   = 4;        // K+1 tokens
constexpr int PP  = 16;       // eig components
constexpr int FFH = 512;
constexpr int NC  = 10;

union FragU { bf16x16 v; uint4 q[2]; };

// A fragment (16x32 bf16, M x K) from LDS, row-major, stride lda.
// lane L: M = L%16 ; elems 0..7 -> K = kb + 8*(L/16) + i ; elems 8..15 -> +16
DEVI bf16x16 frag_A(const __bf16* A, int lda, int kb, int lane) {
  const __bf16* p = A + (lane & 15) * lda + kb + ((lane >> 4) << 3);
  FragU f;
  f.q[0] = *(const uint4*)(p);
  f.q[1] = *(const uint4*)(p + 16);
  return f.v;
}

// B fragment (32x16 bf16, K x N) with B[k][n] = W[n][k], W row-major (Hout x Hin).
// lane L: N = L%16 ; elems 0..7 -> K = kb + 8*(L/16) + i ; elems 8..15 -> +16
DEVI bf16x16 frag_B(const __bf16* W, int ldw, int n0, int kb, int lane) {
  const __bf16* p = W + (long)(n0 + (lane & 15)) * ldw + kb + ((lane >> 4) << 3);
  FragU f;
  f.q[0] = *(const uint4*)(p);
  f.q[1] = *(const uint4*)(p + 16);
  return f.v;
}

DEVI f32x8 wmma_bf16(bf16x16 a, bf16x16 b, f32x8 c) {
  return __builtin_amdgcn_wmma_f32_16x16x32_bf16(false, a, false, b, (short)0, c,
                                                 false, false);
}

DEVI f32x8 fzero8() {
  f32x8 z = {0.f, 0.f, 0.f, 0.f, 0.f, 0.f, 0.f, 0.f};
  return z;
}

DEVI float nan2num(float x) {
  if (__builtin_isnan(x)) return 0.f;
  if (__builtin_isinf(x)) return x > 0.f ? 3.4028234663852886e38f
                                         : -3.4028234663852886e38f;
  return x;
}

#if USE_TDM
// Tensor Data Mover: flat 1-D copy of `elems` f32 from global to LDS.
// D# packing per CDNA5 ISA:
//   group0: count=1 | lds_addr[63:32] | global_addr[120:64] | type=2[127:126]
//   group1: data_size=4B(code 2) | tensor_dim0=elems | tensor_dim1=1 |
//           tile_dim0=elems | tile_dim1=1 | tensor_dim0_stride=elems
DEVI void tdm_load_rows(void* lds_dst, const void* gsrc, unsigned elems) {
  unsigned long long ga = (unsigned long long)(uintptr_t)gsrc;
  unsigned lds = (unsigned)(uintptr_t)lds_dst;
  u32x4 g0;
  g0[0] = 1u;                                              // count=1
  g0[1] = lds;                                             // lds_addr
  g0[2] = (unsigned)(ga & 0xffffffffu);                    // global_addr lo
  g0[3] = (unsigned)((ga >> 32) & 0x01ffffffu) | (2u << 30);  // ga hi | type=2
  i32x8 g1;
  g1[0] = (int)(2u << 16);                                 // data_size = 4 bytes
  g1[1] = (int)((elems & 0xffffu) << 16);                  // tensor_dim0[15:0]
  g1[2] = (int)(((elems >> 16) & 0xffffu) | (1u << 16));   // dim0 hi | tensor_dim1=1
  g1[3] = (int)((elems & 0xffffu) << 16);                  // tile_dim0 (<=65535)
  g1[4] = 1;                                               // tile_dim1 = 1
  g1[5] = (int)elems;                                      // dim0_stride lo32
  g1[6] = 0;
  g1[7] = 0;
  i32x4 z4;
  z4[0] = 0; z4[1] = 0; z4[2] = 0; z4[3] = 0;
#if __clang_major__ >= 23
  i32x8 z8;
  for (int i = 0; i < 8; ++i) z8[i] = 0;
  __builtin_amdgcn_tensor_load_to_lds(g0, g1, z4, z4, z8, 0);
#else
  __builtin_amdgcn_tensor_load_to_lds(g0, g1, z4, z4, 0);
#endif
}
#endif  // USE_TDM

// ---------------------------------------------------------------------------
// Generic fused GEMM: out(16 rows x Hout) = act( preLN(in) @ W^T + bias )
// optional post-LN, optional accumulate into out.  4 waves / 16 rows / block.
// kb-outer loop: one A fragment feeds up to 8 WMMAs.  Wave index is
// scalarized with readfirstlane so all tile guards are scalar branches and
// EXEC stays all-ones around v_wmma (ISA 7.12 requirement).
// ---------------------------------------------------------------------------
__global__ void __launch_bounds__(128)
k_gemm16(const float* __restrict__ in, int ld_in,
         const float* __restrict__ pre_g, const float* __restrict__ pre_b,
         const __bf16* __restrict__ W, int Hin,
         const float* __restrict__ bias,
         float* __restrict__ out, int ld_out, int Hout_pad, int Hout,
         int act /*0 none,1 relu,2 gelu*/, int accumulate,
         const float* __restrict__ post_g, const float* __restrict__ post_b)
{
  __shared__ float  sf[16 * 512];
  __shared__ __bf16 sb[16 * 512];
  __shared__ float  sm[16], srs[16];
  const int tid  = threadIdx.x;
  const int lane = tid & 31;
  const int wave = __builtin_amdgcn_readfirstlane(tid >> 5);  // wave-uniform SGPR
  const long r0  = (long)blockIdx.x * 16;

  // stage 16 rows of A (f32)
  const int nv = Hin >> 2;  // float4 per row
  for (int idx = tid; idx < 16 * nv; idx += 128) {
    int r = idx / nv, cv = idx - r * nv;
    *(float4*)(sf + r * Hin + cv * 4) =
        *(const float4*)(in + (r0 + r) * (long)ld_in + cv * 4);
  }
  __syncthreads();

  if (pre_g) {
    if (tid < 16) {
      float m = 0.f;
      for (int c = 0; c < Hin; ++c) m += sf[tid * Hin + c];
      m *= (1.f / Hin);
      float v = 0.f;
      for (int c = 0; c < Hin; ++c) { float d = sf[tid * Hin + c] - m; v += d * d; }
      v *= (1.f / Hin);
      sm[tid]  = m;
      srs[tid] = rsqrtf(v + 1e-5f);
    }
    __syncthreads();
    for (int idx = tid; idx < 16 * Hin; idx += 128) {
      int r = idx / Hin, c = idx - r * Hin;
      sb[idx] = (__bf16)((sf[idx] - sm[r]) * srs[r] * pre_g[c] + pre_b[c]);
    }
  } else {
    for (int idx = tid; idx < 16 * Hin; idx += 128) sb[idx] = (__bf16)sf[idx];
  }
  __syncthreads();

  // this wave's output-column tiles (scalar)
  const int ntiles = Hout_pad >> 4;
  int n0s[8];
  int mytiles = 0;
  for (int nt = wave; nt < ntiles; nt += 4) n0s[mytiles++] = nt << 4;
  mytiles = __builtin_amdgcn_readfirstlane(mytiles);

  f32x8 acc[8];
#pragma unroll
  for (int t = 0; t < 8; ++t) acc[t] = fzero8();

  for (int kb = 0; kb < Hin; kb += 32) {
    bf16x16 a = frag_A(sb, Hin, kb, lane);
#pragma unroll
    for (int t = 0; t < 8; ++t) {
      if (t < mytiles) {
        if (kb + 32 < Hin)
          __builtin_prefetch(W + (long)(n0s[t] + (lane & 15)) * Hin + kb + 32, 0, 1);
        acc[t] = wmma_bf16(a, frag_B(W, Hin, n0s[t], kb, lane), acc[t]);
      }
    }
  }

  const int n  = lane & 15;
  const int mb = (lane >> 4) << 3;
#pragma unroll
  for (int t = 0; t < 8; ++t) {
    if (t >= mytiles) break;
    const int col = n0s[t] + n;
    const float bv = (bias && col < Hout) ? bias[col] : 0.f;
    for (int r = 0; r < 8; ++r) {
      float v = acc[t][r] + bv;
      if (act == 1) v = fmaxf(v, 0.f);
      else if (act == 2) v = 0.5f * v * (1.f + erff(v * 0.70710678118654752f));
      if (post_g) {
        sf[(mb + r) * Hout_pad + col] = v;
      } else if (col < Hout) {
        float* o = out + (r0 + mb + r) * (long)ld_out + col;
        if (accumulate) *o += v; else *o = v;
      }
    }
  }

  if (post_g) {
    __syncthreads();
    if (tid < 16) {
      float m = 0.f;
      for (int c = 0; c < Hout; ++c) m += sf[tid * Hout_pad + c];
      m *= (1.f / Hout);
      float v = 0.f;
      for (int c = 0; c < Hout; ++c) { float d = sf[tid * Hout_pad + c] - m; v += d * d; }
      v *= (1.f / Hout);
      sm[tid]  = m;
      srs[tid] = rsqrtf(v + 1e-5f);
    }
    __syncthreads();
    for (int idx = tid; idx < 16 * Hout; idx += 128) {
      int r = idx / Hout, c = idx - r * Hout;
      out[(r0 + r) * (long)ld_out + c] =
          (sf[r * Hout_pad + c] - sm[r]) * srs[r] * post_g[c] + post_b[c];
    }
  }
}

// ---------------------------------------------------------------------------
// Fused attention layer: 4 nodes (16 rows) per block.
// tokens += attn(LN1(tokens) @ qkvW^T + b) @ outW^T + ob
// ---------------------------------------------------------------------------
__global__ void __launch_bounds__(128)
k_attn(const float* __restrict__ tokens_in, float* __restrict__ tokens,
       const __bf16* __restrict__ qkvW, const float* __restrict__ qkvB,
       const __bf16* __restrict__ oW,   const float* __restrict__ oB,
       const float* __restrict__ g1,    const float* __restrict__ b1)
{
  __shared__ float  stok[16 * 128];
  __shared__ __bf16 sa[16 * 128];
  __shared__ float  sqkv[16 * 384];
  __shared__ __bf16 so[16 * 128];
  __shared__ float  sm[16], srs[16];
  const int tid  = threadIdx.x;
  const int lane = tid & 31;
  const int wave = __builtin_amdgcn_readfirstlane(tid >> 5);
  const long r0  = (long)blockIdx.x * 16;  // rows (n*4+s) are contiguous

#if USE_TDM
  if (wave == 0) {
    tdm_load_rows(stok, tokens_in + r0 * 128, 16u * 128u);
    __builtin_amdgcn_s_wait_tensorcnt(0);
  }
  __syncthreads();
#else
  for (int idx = tid; idx < 16 * 32; idx += 128) {
    int r = idx >> 5, cv = idx & 31;
    *(float4*)(stok + r * 128 + cv * 4) =
        *(const float4*)(tokens_in + (r0 + r) * 128 + cv * 4);
  }
  __syncthreads();
#endif

  if (tid < 16) {
    float m = 0.f;
    for (int c = 0; c < 128; ++c) m += stok[tid * 128 + c];
    m *= (1.f / 128.f);
    float v = 0.f;
    for (int c = 0; c < 128; ++c) { float d = stok[tid * 128 + c] - m; v += d * d; }
    sm[tid] = m; srs[tid] = rsqrtf(v * (1.f / 128.f) + 1e-5f);
  }
  __syncthreads();
  for (int idx = tid; idx < 2048; idx += 128) {
    int r = idx >> 7, c = idx & 127;
    sa[idx] = (__bf16)((stok[idx] - sm[r]) * srs[r] * g1[c] + b1[c]);
  }
  __syncthreads();

  // QKV GEMM: Hout = 384 -> 24 tiles, 6 per wave; kb-outer so one A frag
  // feeds 6 WMMAs.
  {
    f32x8 acc[6];
#pragma unroll
    for (int t = 0; t < 6; ++t) acc[t] = fzero8();
    for (int kb = 0; kb < 128; kb += 32) {
      bf16x16 a = frag_A(sa, 128, kb, lane);
#pragma unroll
      for (int t = 0; t < 6; ++t)
        acc[t] = wmma_bf16(a, frag_B(qkvW, 128, (wave + 4 * t) << 4, kb, lane),
                           acc[t]);
    }
    const int n = lane & 15, mb = (lane >> 4) << 3;
#pragma unroll
    for (int t = 0; t < 6; ++t) {
      const int col = ((wave + 4 * t) << 4) + n;
      const float bv = qkvB[col];
      for (int r = 0; r < 8; ++r) sqkv[(mb + r) * 384 + col] = acc[t][r] + bv;
    }
  }
  __syncthreads();

  // attention: 4 nodes x 8 heads x 4 query tokens = 128 threads
  {
    const int pr = tid >> 2, sq = tid & 3;
    const int nl = pr >> 3, hd = pr & 7;
    const int qr = nl * 4 + sq;
    const float* qp = sqkv + qr * 384 + hd * 16;
    float sc[4];
    float mx = -1e30f;
    for (int t = 0; t < 4; ++t) {
      const float* kp = sqkv + (nl * 4 + t) * 384 + 128 + hd * 16;
      float d = 0.f;
      for (int i = 0; i < 16; ++i) d += qp[i] * kp[i];
      sc[t] = d * 0.25f;  // 1/sqrt(dh=16)
      mx = fmaxf(mx, sc[t]);
    }
    float sum = 0.f;
    for (int t = 0; t < 4; ++t) { sc[t] = __expf(sc[t] - mx); sum += sc[t]; }
    const float inv = 1.f / sum;
    for (int i = 0; i < 16; ++i) {
      float o = 0.f;
      for (int t = 0; t < 4; ++t)
        o += sc[t] * sqkv[(nl * 4 + t) * 384 + 256 + hd * 16 + i];
      so[qr * 128 + hd * 16 + i] = (__bf16)(o * inv);
    }
  }
  __syncthreads();

  // out-proj GEMM + residual: 8 tiles, 2 per wave
  {
    f32x8 acc[2];
    acc[0] = fzero8(); acc[1] = fzero8();
    for (int kb = 0; kb < 128; kb += 32) {
      bf16x16 a = frag_A(so, 128, kb, lane);
#pragma unroll
      for (int t = 0; t < 2; ++t)
        acc[t] = wmma_bf16(a, frag_B(oW, 128, (wave + 4 * t) << 4, kb, lane),
                           acc[t]);
    }
    const int n = lane & 15, mb = (lane >> 4) << 3;
#pragma unroll
    for (int t = 0; t < 2; ++t) {
      const int col = ((wave + 4 * t) << 4) + n;
      const float bv = oB[col];
      for (int r = 0; r < 8; ++r)
        tokens[(r0 + mb + r) * 128 + col] =
            stok[(mb + r) * 128 + col] + acc[t][r] + bv;
    }
  }
}

// ---------------------------------------------------------------------------
// Fused FFN layer: tokens += relu(LN2(tokens)@W1^T + b1) @ W2^T + b2
// ---------------------------------------------------------------------------
__global__ void __launch_bounds__(128)
k_ffn(float* __restrict__ tokens,
      const __bf16* __restrict__ W1, const float* __restrict__ B1,
      const __bf16* __restrict__ W2, const float* __restrict__ B2,
      const float* __restrict__ g2, const float* __restrict__ b2)
{
  __shared__ float  stok[16 * 128];
  __shared__ __bf16 sa[16 * 128];
  __shared__ __bf16 st1[16 * 512];
  __shared__ float  sm[16], srs[16];
  const int tid  = threadIdx.x;
  const int lane = tid & 31;
  const int wave = __builtin_amdgcn_readfirstlane(tid >> 5);
  const long r0  = (long)blockIdx.x * 16;

#if USE_TDM
  if (wave == 0) {
    tdm_load_rows(stok, tokens + r0 * 128, 16u * 128u);
    __builtin_amdgcn_s_wait_tensorcnt(0);
  }
  __syncthreads();
#else
  for (int idx = tid; idx < 16 * 32; idx += 128) {
    int r = idx >> 5, cv = idx & 31;
    *(float4*)(stok + r * 128 + cv * 4) =
        *(const float4*)(tokens + (r0 + r) * 128 + cv * 4);
  }
  __syncthreads();
#endif

  if (tid < 16) {
    float m = 0.f;
    for (int c = 0; c < 128; ++c) m += stok[tid * 128 + c];
    m *= (1.f / 128.f);
    float v = 0.f;
    for (int c = 0; c < 128; ++c) { float d = stok[tid * 128 + c] - m; v += d * d; }
    sm[tid] = m; srs[tid] = rsqrtf(v * (1.f / 128.f) + 1e-5f);
  }
  __syncthreads();
  for (int idx = tid; idx < 2048; idx += 128) {
    int r = idx >> 7, c = idx & 127;
    sa[idx] = (__bf16)((stok[idx] - sm[r]) * srs[r] * g2[c] + b2[c]);
  }
  __syncthreads();

  // GEMM1: 128 -> 512, relu, straight to bf16 LDS; 8 tiles per wave, kb-outer
  {
    f32x8 acc[8];
#pragma unroll
    for (int t = 0; t < 8; ++t) acc[t] = fzero8();
    for (int kb = 0; kb < 128; kb += 32) {
      bf16x16 a = frag_A(sa, 128, kb, lane);
#pragma unroll
      for (int t = 0; t < 8; ++t)
        acc[t] = wmma_bf16(a, frag_B(W1, 128, (wave + 4 * t) << 4, kb, lane),
                           acc[t]);
    }
    const int n = lane & 15, mb = (lane >> 4) << 3;
#pragma unroll
    for (int t = 0; t < 8; ++t) {
      const int col = ((wave + 4 * t) << 4) + n;
      const float bv = B1[col];
      for (int r = 0; r < 8; ++r)
        st1[(mb + r) * 512 + col] = (__bf16)fmaxf(acc[t][r] + bv, 0.f);
    }
  }
  __syncthreads();

  // GEMM2: 512 -> 128 + residual; 2 tiles per wave, 16 K-steps
  {
    f32x8 acc[2];
    acc[0] = fzero8(); acc[1] = fzero8();
    for (int kb = 0; kb < 512; kb += 32) {
      bf16x16 a = frag_A(st1, 512, kb, lane);
#pragma unroll
      for (int t = 0; t < 2; ++t) {
        const int n0 = (wave + 4 * t) << 4;
        if (kb + 32 < 512)
          __builtin_prefetch(W2 + (long)(n0 + (lane & 15)) * 512 + kb + 32, 0, 1);
        acc[t] = wmma_bf16(a, frag_B(W2, 512, n0, kb, lane), acc[t]);
      }
    }
    const int n = lane & 15, mb = (lane >> 4) << 3;
#pragma unroll
    for (int t = 0; t < 2; ++t) {
      const int col = ((wave + 4 * t) << 4) + n;
      const float bv = B2[col];
      for (int r = 0; r < 8; ++r)
        tokens[(r0 + mb + r) * 128 + col] =
            stok[(mb + r) * 128 + col] + acc[t][r] + bv;
    }
  }
}

// --------------------------- graph / PE helpers ----------------------------

__global__ void k_cvt(const float* __restrict__ s, __bf16* __restrict__ d, int n) {
  int i = blockIdx.x * 256 + threadIdx.x;
  if (i < n) d[i] = (__bf16)s[i];
}

__global__ void k_pad_w2(const float* __restrict__ s, __bf16* __restrict__ d) {
  int i = blockIdx.x * 256 + threadIdx.x;
  if (i < 16 * 128) d[i] = (i >> 7) < NC ? (__bf16)s[i] : (__bf16)0.f;
}

__global__ void k_zero_hops(float* __restrict__ tokens) {
  long i = (long)blockIdx.x * 256 + threadIdx.x;
  if (i < (long)NN * 3 * H) {
    long n = i / (3 * H);
    long rem = i - n * (3 * H);
    tokens[n * (S * H) + H + rem] = 0.f;
  }
}

__global__ void k_deg(const long long* __restrict__ col, float* __restrict__ deg) {
  int e = blockIdx.x * 256 + threadIdx.x;
  if (e < EE) atomicAdd(&deg[col[e]], 1.f);
}

__global__ void k_dinv(const float* __restrict__ deg, float* __restrict__ dinv) {
  int n = blockIdx.x * 256 + threadIdx.x;
  if (n < NN) dinv[n] = rsqrtf(fmaxf(deg[n], 1.f));
}

// one hop: tout[row] += dinv[row]*dinv[col] * tin[col]; strides of 512 floats/node
__global__ void __launch_bounds__(256)
k_spmm(const long long* __restrict__ row, const long long* __restrict__ col,
       const float* __restrict__ dinv,
       const float* __restrict__ tin, float* __restrict__ tout)
{
  long t = (long)blockIdx.x * 256 + threadIdx.x;
  long e = t >> 5;
  if (e >= EE) return;
  int c4 = (int)(t & 31) << 2;
  long r = row[e], c = col[e];
  float nm = dinv[r] * dinv[c];
  float4 v = *(const float4*)(tin + c * (S * H) + c4);
  float* d = tout + r * (S * H) + c4;
  atomicAdd(d + 0, nm * v.x);
  atomicAdd(d + 1, nm * v.y);
  atomicAdd(d + 2, nm * v.z);
  atomicAdd(d + 3, nm * v.w);
}

// u[n][h] = sum_p relu(w1[h][0]*vec + w1[h][1]*(val+eps))  (phi1 + relu + P-sum)
__global__ void __launch_bounds__(128)
k_pe_u(const float* __restrict__ eigvecs, const float* __restrict__ eigvals,
       const float* __restrict__ pe_eps, const float* __restrict__ w1,
       float* __restrict__ u)
{
  __shared__ float se[PP], sv[PP];
  const int n = blockIdx.x, h = threadIdx.x;
  if (h < PP) {
    se[h] = nan2num(eigvecs[(long)n * PP + h]);
    sv[h] = nan2num(eigvals[(long)n * PP + h] + pe_eps[h]);
  }
  __syncthreads();
  const float w0 = w1[h * 2 + 0], wA = w1[h * 2 + 1];
  float acc = 0.f;
  for (int p = 0; p < PP; ++p) acc += fmaxf(w0 * se[p] + wA * sv[p], 0.f);
  u[(long)n * H + h] = acc;
}

__global__ void k_addemb(float* __restrict__ tokens, const float* __restrict__ emb) {
  long i = (long)blockIdx.x * 256 + threadIdx.x;
  if (i < (long)NN * S * H) tokens[i] += emb[i & (S * H - 1)];
}

}  // namespace

// ---------------------------------------------------------------------------
extern "C" void kernel_launch(void* const* d_in, const int* in_sizes, int n_in,
                              void* d_out, int out_size, void* d_ws, size_t ws_size,
                              hipStream_t stream)
{
  (void)in_sizes; (void)n_in; (void)out_size; (void)ws_size;
  const float*     x        = (const float*)d_in[0];
  const long long* eidx     = (const long long*)d_in[1];
  const float*     eigvecs  = (const float*)d_in[2];
  const float*     eigvals  = (const float*)d_in[3];
  const float*     enc_w    = (const float*)d_in[4];
  const float*     enc_b    = (const float*)d_in[5];
  const float*     in_ln_g  = (const float*)d_in[6];
  const float*     in_ln_b  = (const float*)d_in[7];
  const float*     phi_w1   = (const float*)d_in[8];
  const float*     phi_w2   = (const float*)d_in[9];
  const float*     rho_w1   = (const float*)d_in[10];
  const float*     rho_b1   = (const float*)d_in[11];
  const float*     rho_w2   = (const float*)d_in[12];
  const float*     rho_b2   = (const float*)d_in[13];
  const float*     pe_eps   = (const float*)d_in[14];
  const float*     hop_emb  = (const float*)d_in[15];
  const float*     qkv_w    = (const float*)d_in[16];
  const float*     qkv_b    = (const float*)d_in[17];
  const float*     out_w    = (const float*)d_in[18];
  const float*     out_b    = (const float*)d_in[19];
  const float*     ln1_g    = (const float*)d_in[20];
  const float*     ln1_b    = (const float*)d_in[21];
  const float*     ln2_g    = (const float*)d_in[22];
  const float*     ln2_b    = (const float*)d_in[23];
  const float*     ff1_w    = (const float*)d_in[24];
  const float*     ff1_b    = (const float*)d_in[25];
  const float*     ff2_w    = (const float*)d_in[26];
  const float*     ff2_b    = (const float*)d_in[27];
  const float*     fin_ln_g = (const float*)d_in[28];
  const float*     fin_ln_b = (const float*)d_in[29];
  const float*     head_w1  = (const float*)d_in[30];
  const float*     head_b1  = (const float*)d_in[31];
  const float*     head_lng = (const float*)d_in[32];
  const float*     head_lnb = (const float*)d_in[33];
  const float*     head_w2  = (const float*)d_in[34];
  const float*     head_b2  = (const float*)d_in[35];
  float* outp = (float*)d_out;

  // ---- workspace carve ----
  uint8_t* wp = (uint8_t*)d_ws;
  auto take = [&](size_t bytes) {
    uint8_t* p = wp;
    wp += (bytes + 255) & ~(size_t)255;
    return p;
  };
  float* tokens = (float*)take((size_t)NN * S * H * 4);
  float* bufA   = (float*)take((size_t)NN * H * 4);
  float* bufB   = (float*)take((size_t)NN * H * 4);
  float* deg    = (float*)take((size_t)NN * 4);
  float* dinv   = (float*)take((size_t)NN * 4);
  __bf16* enc_bf  = (__bf16*)take(H * H * 2);
  __bf16* pw2_bf  = (__bf16*)take(H * H * 2);
  __bf16* rw1_bf  = (__bf16*)take(H * H * 2);
  __bf16* rw2_bf  = (__bf16*)take(H * H * 2);
  __bf16* qkv_bf  = (__bf16*)take(2 * 3 * H * H * 2);
  __bf16* ow_bf   = (__bf16*)take(2 * H * H * 2);
  __bf16* ff1_bf  = (__bf16*)take(2 * FFH * H * 2);
  __bf16* ff2_bf  = (__bf16*)take(2 * H * FFH * 2);
  __bf16* hw1_bf  = (__bf16*)take(H * H * 2);
  __bf16* hw2_bf  = (__bf16*)take(16 * H * 2);

  auto cvt = [&](const float* s, __bf16* d, int n) {
    k_cvt<<<(n + 255) / 256, 256, 0, stream>>>(s, d, n);
  };
  cvt(enc_w,  enc_bf, H * H);
  cvt(phi_w2, pw2_bf, H * H);
  cvt(rho_w1, rw1_bf, H * H);
  cvt(rho_w2, rw2_bf, H * H);
  cvt(qkv_w,  qkv_bf, 2 * 3 * H * H);
  cvt(out_w,  ow_bf,  2 * H * H);
  cvt(ff1_w,  ff1_bf, 2 * FFH * H);
  cvt(ff2_w,  ff2_bf, 2 * H * FFH);
  cvt(head_w1, hw1_bf, H * H);
  k_pad_w2<<<8, 256, 0, stream>>>(head_w2, hw2_bf);

  // ---- encoder: tokens[:,0,:] = LN(x @ enc_w^T + enc_b) ----
  k_gemm16<<<NN / 16, 128, 0, stream>>>(x, H, nullptr, nullptr, enc_bf, H, enc_b,
                                        tokens, S * H, H, H, 0, 0, in_ln_g, in_ln_b);

  // ---- hop propagation ----
  k_zero_hops<<<(int)(((long)NN * 3 * H + 255) / 256), 256, 0, stream>>>(tokens);
  hipMemsetAsync(deg, 0, (size_t)NN * 4, stream);
  k_deg<<<(EE + 255) / 256, 256, 0, stream>>>(eidx + EE, deg);
  k_dinv<<<(NN + 255) / 256, 256, 0, stream>>>(deg, dinv);
  const int spmm_blocks = (int)(((long)EE * 32 + 255) / 256);
  for (int s = 1; s <= 3; ++s)
    k_spmm<<<spmm_blocks, 256, 0, stream>>>(eidx, eidx + EE, dinv,
                                            tokens + (s - 1) * H, tokens + s * H);

  // ---- positional encoding ----
  k_pe_u<<<NN, 128, 0, stream>>>(eigvecs, eigvals, pe_eps, phi_w1, bufA);
  k_gemm16<<<NN / 16, 128, 0, stream>>>(bufA, H, nullptr, nullptr, pw2_bf, H, nullptr,
                                        bufB, H, H, H, 0, 0, nullptr, nullptr);
  k_gemm16<<<NN / 16, 128, 0, stream>>>(bufB, H, nullptr, nullptr, rw1_bf, H, rho_b1,
                                        bufA, H, H, H, 1, 0, nullptr, nullptr);
  k_gemm16<<<NN / 16, 128, 0, stream>>>(bufA, H, nullptr, nullptr, rw2_bf, H, rho_b2,
                                        tokens, S * H, H, H, 0, 1, nullptr, nullptr);
  k_addemb<<<(int)(((long)NN * S * H + 255) / 256), 256, 0, stream>>>(tokens, hop_emb);

  // ---- transformer layers ----
  const int rows_blocks = NN * S / 16;  // 25000
  for (int l = 0; l < 2; ++l) {
    k_attn<<<rows_blocks, 128, 0, stream>>>(tokens, tokens,
                                            qkv_bf + (size_t)l * 3 * H * H,
                                            qkv_b + l * 3 * H,
                                            ow_bf + (size_t)l * H * H,
                                            out_b + l * H,
                                            ln1_g + l * H, ln1_b + l * H);
    k_ffn<<<rows_blocks, 128, 0, stream>>>(tokens,
                                           ff1_bf + (size_t)l * FFH * H,
                                           ff1_b + l * FFH,
                                           ff2_bf + (size_t)l * H * FFH,
                                           ff2_b + l * H,
                                           ln2_g + l * H, ln2_b + l * H);
  }

  // ---- head (token 0 only): gelu(LN_fin(t0)@W1^T+b1) -> LN_head -> W2 ----
  k_gemm16<<<NN / 16, 128, 0, stream>>>(tokens, S * H, fin_ln_g, fin_ln_b,
                                        hw1_bf, H, head_b1,
                                        bufA, H, H, H, 2, 0, nullptr, nullptr);
  k_gemm16<<<NN / 16, 128, 0, stream>>>(bufA, H, head_lng, head_lnb,
                                        hw2_bf, H, head_b2,
                                        outp, NC, 16, NC, 0, 0, nullptr, nullptr);
}
